// SoftMoE_55654186221571
// MI455X (gfx1250) — matrix operand order repaired and checked by reading, not compiled, not run on hardware
//
#include <hip/hip_runtime.h>

// ---------------- problem constants ----------------
#define Bdim 8
#define Mdim 8192
#define Ddim 768
#define NE   64
#define SL   2
#define NP   128   // NE*SL
#define DOUT 768

// ---------------- WMMA helper types ----------------
typedef __attribute__((ext_vector_type(16))) __bf16          v16bf;
typedef __attribute__((ext_vector_type(16))) unsigned short  v16us;
typedef __attribute__((ext_vector_type(8)))  float           v8f;

union BF16x16 { v16bf bf; v16us us; uint4 q[2]; };

__device__ __forceinline__ unsigned short f2bf(float f) {
    unsigned int u = __float_as_uint(f);
    u += 0x7FFFu + ((u >> 16) & 1u);          // round-to-nearest-even
    return (unsigned short)(u >> 16);
}

__device__ __forceinline__ v8f vzero8() {
    v8f v;
#pragma unroll
    for (int i = 0; i < 8; ++i) v[i] = 0.0f;
    return v;
}

__device__ __forceinline__ v8f wmma_bf16(v16bf a, v16bf b, v8f c) {
    return __builtin_amdgcn_wmma_f32_16x16x32_bf16(
        false, a, false, b, (short)0, c, false, false);
}

// A tile 16x32 (MxK), bf16 row-major source, per ISA 16-bit A layout:
// lane<16 : K = {0..7,16..23}; lane>=16 : K = {8..15,24..31}
__device__ __forceinline__ v16bf load_A_bf16(const unsigned short* a, int lda) {
    int l = threadIdx.x & 31, row = l & 15, kh = l >> 4;
    const unsigned short* p = a + row * lda + kh * 8;
    BF16x16 r;
    r.q[0] = *(const uint4*)(p);
    r.q[1] = *(const uint4*)(p + 16);
    return r.bf;
}

// Same A layout, loading f32 and converting to bf16 in-register.
__device__ __forceinline__ v16bf load_A_f32(const float* a, int lda) {
    int l = threadIdx.x & 31, row = l & 15, kh = l >> 4;
    const float* p = a + row * lda + kh * 8;
    float4 f0 = *(const float4*)(p);
    float4 f1 = *(const float4*)(p + 4);
    float4 f2 = *(const float4*)(p + 16);
    float4 f3 = *(const float4*)(p + 20);
    float t[16] = {f0.x, f0.y, f0.z, f0.w, f1.x, f1.y, f1.z, f1.w,
                   f2.x, f2.y, f2.z, f2.w, f3.x, f3.y, f3.z, f3.w};
    BF16x16 r;
#pragma unroll
    for (int i = 0; i < 16; ++i) r.us[i] = f2bf(t[i]);
    return r.bf;
}

// B tile 32x16 (KxN) from TRANSPOSED storage bt[N][K] row-major:
// lane<16 : K = {0..7,16..23}; lane>=16 : K = {8..15,24..31}, N = lane&15.
__device__ __forceinline__ v16bf load_B_bt(const unsigned short* bt, int ldk) {
    int l = threadIdx.x & 31, n = l & 15, kh = l >> 4;
    const unsigned short* p = bt + n * ldk + kh * 8;
    BF16x16 r;
    r.q[0] = *(const uint4*)(p);
    r.q[1] = *(const uint4*)(p + 16);
    return r.bf;
}

// ---------------- prep kernels ----------------

// phiT[np][d] = bf16(phi[d][np])
__global__ void k_phiT(const float* __restrict__ phi, unsigned short* __restrict__ phiT) {
    int id = blockIdx.x * blockDim.x + threadIdx.x;   // NP*Ddim threads
    int np = id / Ddim, d = id % Ddim;
    phiT[np * Ddim + d] = f2bf(phi[d * NP + np]);
}

// WT[n][d][k] = bf16(W[n][k][d])  (LDS-tiled 32x32 transpose)
__global__ void k_wt(const float* __restrict__ W, unsigned short* __restrict__ WT) {
    __shared__ float t[32][33];
    int n = blockIdx.z;
    int k0 = blockIdx.x * 32, d0 = blockIdx.y * 32;
    const float* Wn = W + (size_t)n * Ddim * DOUT;
    for (int i = threadIdx.y; i < 32; i += 8)
        t[i][threadIdx.x] = Wn[(size_t)(k0 + i) * DOUT + d0 + threadIdx.x];
    __syncthreads();
    unsigned short* WTn = WT + (size_t)n * DOUT * Ddim;
    for (int i = threadIdx.y; i < 32; i += 8)
        WTn[(size_t)(d0 + i) * Ddim + k0 + threadIdx.x] = f2bf(t[threadIdx.x][i]);
}

// ---------------- stage 1: logits = x @ phi  (WMMA bf16) ----------------
// one wave -> one 16(m) x 128(np) tile, K = 768.
// All 8 B tiles issued first; A f32->bf16 conversion VALU hides their latency.
__global__ void __launch_bounds__(128)
k_logits(const float* __restrict__ x, const unsigned short* __restrict__ phiT,
         float* __restrict__ logits) {
    int tile = blockIdx.x * 4 + (threadIdx.x >> 5);     // 0..4095
    int b = tile >> 9;
    int m0 = (tile & 511) << 4;
    const float* xa = x + ((size_t)b * Mdim + m0) * Ddim;

    v8f acc[8];
#pragma unroll
    for (int i = 0; i < 8; ++i) acc[i] = vzero8();

    for (int k0 = 0; k0 < Ddim; k0 += 32) {
        v16bf bt[8];
#pragma unroll
        for (int nt = 0; nt < 8; ++nt)
            bt[nt] = load_B_bt(phiT + (nt * 16) * Ddim + k0, Ddim);
        v16bf a = load_A_f32(xa + k0, Ddim);
#pragma unroll
        for (int nt = 0; nt < 8; ++nt)
            acc[nt] = wmma_bf16(a, bt[nt], acc[nt]);
    }
    int l = threadIdx.x & 31, col = l & 15, kh = l >> 4;
#pragma unroll
    for (int nt = 0; nt < 8; ++nt)
#pragma unroll
        for (int v = 0; v < 8; ++v)
            logits[((size_t)b * Mdim + m0 + v + 8 * kh) * NP + nt * 16 + col] = acc[nt][v];
}

// ---------------- stage 2: dual softmax ----------------
// Bulk-stages 32KB of logits into LDS with async global->LDS copies (ASYNCcnt),
// then: dispatch = softmax over b -> n-summed dsum;  combine = softmax over np (bf16).
__global__ void __launch_bounds__(256)
k_softmax(const float* __restrict__ logits, unsigned short* __restrict__ comb,
          float* __restrict__ dsum) {
    __shared__ float L[Bdim][8][NP];      // 32 KB, L[b][i][np]
    __shared__ float DS[Bdim][8][SL];
    int m0 = blockIdx.x * 8;
    int tid = threadIdx.x;

    if (tid < 128) DS[tid >> 4][(tid >> 1) & 7][tid & 1] = 0.0f;

    // async copy: 2048 x 16B transfers; LDS is b-major so each b is one
    // contiguous 4KB span of global memory (rows m0..m0+7).
    {
        unsigned lds_base = (unsigned)(size_t)(void*)&L[0][0][0];
#pragma unroll
        for (int q = 0; q < 8; ++q) {
            int idx = tid + 256 * q;                       // 0..2047
            int b = idx >> 8;                              // 256 x16B per b
            unsigned lds_addr = lds_base + (unsigned)idx * 16u;
            unsigned long long gaddr = (unsigned long long)(size_t)
                (logits + ((size_t)b * Mdim + m0) * NP) + (unsigned)(idx & 255) * 16u;
            asm volatile("global_load_async_to_lds_b128 %0, %1, off"
                         :: "v"(lds_addr), "v"(gaddr) : "memory");
        }
        asm volatile("s_wait_asynccnt 0x0" ::: "memory");
    }
    __syncthreads();

    // dispatch: softmax over b for each (i,np); accumulate n-sum into DS via LDS atomics
#pragma unroll
    for (int q = 0; q < 4; ++q) {
        int it = tid + 256 * q;           // 1024 items
        int i = it >> 7, np = it & 127, p = np & 1;
        float v[Bdim], mx = -3.0e38f;
#pragma unroll
        for (int b = 0; b < Bdim; ++b) { v[b] = L[b][i][np]; mx = fmaxf(mx, v[b]); }
        float s = 0.0f;
#pragma unroll
        for (int b = 0; b < Bdim; ++b) { v[b] = __expf(v[b] - mx); s += v[b]; }
        float inv = 1.0f / s;
#pragma unroll
        for (int b = 0; b < Bdim; ++b) atomicAdd(&DS[b][i][p], v[b] * inv);
    }

    // combine: softmax over np (row of 128) per (b,i); wave32 shfl reductions
    int wave = tid >> 5, lane = tid & 31;
#pragma unroll
    for (int rr = 0; rr < 8; ++rr) {
        int r = wave * 8 + rr, b = r >> 3, i = r & 7;
        float v[4], mx = -3.0e38f;
#pragma unroll
        for (int j = 0; j < 4; ++j) { v[j] = L[b][i][lane + 32 * j]; mx = fmaxf(mx, v[j]); }
#pragma unroll
        for (int off = 16; off >= 1; off >>= 1) mx = fmaxf(mx, __shfl_xor(mx, off, 32));
        float s = 0.0f;
#pragma unroll
        for (int j = 0; j < 4; ++j) { v[j] = __expf(v[j] - mx); s += v[j]; }
#pragma unroll
        for (int off = 16; off >= 1; off >>= 1) s += __shfl_xor(s, off, 32);
        float inv = 1.0f / s;
        unsigned short* crow = comb + ((size_t)b * Mdim + m0 + i) * NP;
#pragma unroll
        for (int j = 0; j < 4; ++j) crow[lane + 32 * j] = f2bf(v[j] * inv);
    }
    __syncthreads();
    if (tid < 128) {
        int b = tid >> 4, i = (tid >> 1) & 7, p = tid & 1;
        dsum[((size_t)b * Mdim + m0 + i) * SL + p] = DS[b][i][p];
    }
}

// ---------------- stage 3: s_sum[b,p,d] = sum_m x[b,m,d]*dsum[b,m,p] ----------------
__global__ void __launch_bounds__(256)
k_part(const float* __restrict__ x, const float* __restrict__ dsum,
       float* __restrict__ part) {
    int d = blockIdx.x * 256 + threadIdx.x;   // 3 x 256 = 768
    int chunk = blockIdx.y;                   // 32 chunks of 256 m
    int b = blockIdx.z;
    float a0 = 0.0f, a1 = 0.0f;
    int mbase = chunk * 256;
    const float* xb = x + ((size_t)b * Mdim + mbase) * Ddim + d;
    const float* db = dsum + ((size_t)b * Mdim + mbase) * SL;
    for (int m = 0; m < 256; ++m) {
        float xv = xb[(size_t)m * Ddim];
        a0 += xv * db[m * SL + 0];
        a1 += xv * db[m * SL + 1];
    }
    part[(((size_t)b * 32 + chunk) * SL + 0) * Ddim + d] = a0;
    part[(((size_t)b * 32 + chunk) * SL + 1) * Ddim + d] = a1;
}

// reduce 32 chunks -> ssum bf16 [16 rows = b*2+p][768]
__global__ void k_red(const float* __restrict__ part, unsigned short* __restrict__ ssum) {
    int id = blockIdx.x * blockDim.x + threadIdx.x;   // 16*768
    int r = id / Ddim, d = id % Ddim;
    int b = r >> 1, p = r & 1;
    float s = 0.0f;
    for (int c = 0; c < 32; ++c)
        s += part[(((size_t)b * 32 + c) * SL + p) * Ddim + d];
    ssum[r * Ddim + d] = f2bf(s);
}

// ---------------- stage 4: y[n] = ssum @ W[n] + bias[n]  (WMMA bf16) ----------------
// one wave -> 16(b,p) x 64(d: 4 tiles) for one expert; A tile feeds 4 WMMAs.
__global__ void __launch_bounds__(256)
k_y(const unsigned short* __restrict__ ssum, const unsigned short* __restrict__ WT,
    const float* __restrict__ bias, unsigned short* __restrict__ yT) {
    int w = blockIdx.x * 8 + (threadIdx.x >> 5);     // 0..767
    int n = w / 12, dg = w % 12;                     // dg = group of 4 d-tiles
    const unsigned short* Bt0 = WT + (size_t)n * DOUT * Ddim + (size_t)(dg * 64) * Ddim;

    v8f acc[4];
#pragma unroll
    for (int t = 0; t < 4; ++t) acc[t] = vzero8();

    for (int k0 = 0; k0 < Ddim; k0 += 32) {
        v16bf bt[4];
#pragma unroll
        for (int t = 0; t < 4; ++t)
            bt[t] = load_B_bt(Bt0 + (size_t)(t * 16) * Ddim + k0, Ddim);
        v16bf a = load_A_bf16(ssum + k0, Ddim);
#pragma unroll
        for (int t = 0; t < 4; ++t)
            acc[t] = wmma_bf16(a, bt[t], acc[t]);
    }
    int l = threadIdx.x & 31, col = l & 15, kh = l >> 4;
#pragma unroll
    for (int t = 0; t < 4; ++t) {
        int d = dg * 64 + t * 16 + col;
        float bv = bias[n * DOUT + d];
#pragma unroll
        for (int v = 0; v < 8; ++v) {
            int row = v + 8 * kh;          // row = b*2 + p
            int b = row >> 1, p = row & 1;
            yT[((size_t)b * DOUT + d) * NP + n * SL + p] = f2bf(acc[t][v] + bv);
        }
    }
}

// ---------------- stage 5: out = combine @ y  (WMMA bf16, K=128) ----------------
// one wave -> 16(m) x 64(d: 4 tiles); A tile reused across 4 WMMAs per k-step.
__global__ void __launch_bounds__(128)
k_out(const unsigned short* __restrict__ comb, const unsigned short* __restrict__ yT,
      float* __restrict__ out) {
    int w = blockIdx.x * 4 + (threadIdx.x >> 5);     // 0..49151
    int dg = w % 12;
    int mt = w / 12;
    int b = mt >> 9;
    int m0 = (mt & 511) << 4;
    const unsigned short* Am  = comb + ((size_t)b * Mdim + m0) * NP;
    const unsigned short* Bt0 = yT + (size_t)b * DOUT * NP + (size_t)(dg * 64) * NP;

    v8f acc[4];
#pragma unroll
    for (int t = 0; t < 4; ++t) acc[t] = vzero8();

#pragma unroll
    for (int k0 = 0; k0 < NP; k0 += 32) {
        v16bf bt[4];
#pragma unroll
        for (int t = 0; t < 4; ++t)
            bt[t] = load_B_bt(Bt0 + (size_t)(t * 16) * NP + k0, NP);
        v16bf a = load_A_bf16(Am + k0, NP);
#pragma unroll
        for (int t = 0; t < 4; ++t)
            acc[t] = wmma_bf16(a, bt[t], acc[t]);
    }
    int l = threadIdx.x & 31, col = l & 15, kh = l >> 4;
#pragma unroll
    for (int t = 0; t < 4; ++t)
#pragma unroll
        for (int v = 0; v < 8; ++v)
            out[((size_t)b * Mdim + m0 + v + 8 * kh) * DOUT + dg * 64 + t * 16 + col]
                = acc[t][v];
}

// ---------------- launcher ----------------
extern "C" void kernel_launch(void* const* d_in, const int* in_sizes, int n_in,
                              void* d_out, int out_size, void* d_ws, size_t ws_size,
                              hipStream_t stream) {
    (void)in_sizes; (void)n_in; (void)out_size; (void)ws_size;
    const float* x    = (const float*)d_in[0];
    const float* phi  = (const float*)d_in[1];
    const float* W    = (const float*)d_in[2];
    const float* bias = (const float*)d_in[3];
    float* out        = (float*)d_out;

    char* ws = (char*)d_ws;
    size_t off = 0;
    auto take = [&](size_t bytes) -> void* {
        void* p = ws + off;
        off += (bytes + 255) & ~(size_t)255;
        return p;
    };
    unsigned short* phiT = (unsigned short*)take((size_t)NP * Ddim * 2);          // 192 KB
    unsigned short* WT   = (unsigned short*)take((size_t)NE * DOUT * Ddim * 2);   // 75.5 MB
    float*          logi = (float*)take((size_t)Bdim * Mdim * NP * 4);            // 33.6 MB
    unsigned short* comb = (unsigned short*)take((size_t)Bdim * Mdim * NP * 2);   // 16.8 MB
    float*          dsum = (float*)take((size_t)Bdim * Mdim * SL * 4);            // 0.5 MB
    float*          part = (float*)take((size_t)Bdim * 32 * SL * Ddim * 4);       // 1.6 MB
    unsigned short* ssum = (unsigned short*)take((size_t)16 * Ddim * 2);          // 24 KB
    unsigned short* yT   = (unsigned short*)take((size_t)Bdim * DOUT * NP * 2);   // 1.6 MB

    k_phiT   <<<(NP * Ddim) / 256, 256, 0, stream>>>(phi, phiT);
    k_wt     <<<dim3(24, 24, NE), dim3(32, 8), 0, stream>>>(W, WT);
    k_logits <<<(Bdim * Mdim / 16) / 4, 128, 0, stream>>>(x, phiT, logi);
    k_softmax<<<Mdim / 8, 256, 0, stream>>>(logi, comb, dsum);
    k_part   <<<dim3(3, 32, Bdim), 256, 0, stream>>>(x, dsum, part);
    k_red    <<<(16 * Ddim) / 256, 256, 0, stream>>>(part, ssum);
    k_y      <<<(NE * 48) / 8 / 4, 256, 0, stream>>>(ssum, WT, bias, yT);
    k_out    <<<((Bdim * Mdim / 16) * 48) / 4 / 4, 128, 0, stream>>>(comb, yT, out);
}
